// TwoWayHydraAttention_2516850835589
// MI455X (gfx1250) — matrix operand order repaired and checked by reading, not compiled, not run on hardware
//
#include <hip/hip_runtime.h>

typedef __bf16  bf16_t;
typedef bf16_t  v8bf  __attribute__((ext_vector_type(8)));
typedef bf16_t  v16bf __attribute__((ext_vector_type(16)));
typedef float   v4f   __attribute__((ext_vector_type(4)));
typedef float   v8f   __attribute__((ext_vector_type(8)));

#define B_SZ   4
#define N_TOK  4096
#define D_DIM  1024
#define KV_DIM 2048

// ---------------------------------------------------------------------------
// bf16 WMMA: D(16x16,f32) += A(16x32,bf16) x B(32x16,bf16)
// Lane L = (l = L%16, half = L/16):
//   A (v16bf): row m0+l, elems 0..7  -> K = k0 + 8*half + 0..7
//                        elems 8..15 -> K = k0 + 16 + 8*half + 0..7
//   B (v16bf): W-row n0+l, elems 0..15 -> same K as A elems 0..15
//   D: lanes<16 -> rows m0+r, lanes>=16 -> rows m0+8+r, col n0 + L%16
// f32 GEMM emulated as hi*hi + hi*lo + lo*hi (2-way bf16 split, f32 accum):
// 3 WMMA issues per K=32 vs 8 issues for V_WMMA_F32_16X16X4_F32 -> ~2.7x.
// ---------------------------------------------------------------------------
__device__ __forceinline__ v8f wmma_bf16(v16bf a, v16bf b, v8f c) {
    return __builtin_amdgcn_wmma_f32_16x16x32_bf16(
        /*neg_a=*/false, a, /*neg_b=*/false, b,
        /*c_mod=*/(short)0, c, /*reuse_a=*/false, /*reuse_b=*/false);
}

// bf16 chunks p[0..7] and p[16..23] -> v16bf (two b128 loads)
__device__ __forceinline__ v16bf load16(const bf16_t* __restrict__ p) {
    v8bf c0 = *(const v8bf*)(p);
    v8bf c1 = *(const v8bf*)(p + 16);
    return __builtin_shufflevector(c0, c1, 0, 1, 2, 3, 4, 5, 6, 7,
                                   8, 9, 10, 11, 12, 13, 14, 15);
}

// A operand from pre-split bf16 buffer: chunks at p[0..7] and p[16..23]
__device__ __forceinline__ v16bf load_a16(const bf16_t* __restrict__ p) {
    v8bf c0 = *(const v8bf*)(p);
    v8bf c1 = *(const v8bf*)(p + 16);
    return __builtin_shufflevector(c0, c1, 0, 1, 2, 3, 4, 5, 6, 7,
                                   8, 9, 10, 11, 12, 13, 14, 15);
}

// A operand split on the fly from f32: chunks p[0..7], p[16..23]
__device__ __forceinline__ void split_a(const float* __restrict__ p,
                                        v16bf& hi, v16bf& lo) {
    v4f q0 = *(const v4f*)(p);
    v4f q1 = *(const v4f*)(p + 4);
    v4f q2 = *(const v4f*)(p + 16);
    v4f q3 = *(const v4f*)(p + 20);
    float v[16];
#pragma unroll
    for (int i = 0; i < 4; ++i) {
        v[i] = q0[i]; v[4 + i] = q1[i]; v[8 + i] = q2[i]; v[12 + i] = q3[i];
    }
#pragma unroll
    for (int i = 0; i < 16; ++i) {
        bf16_t h = (bf16_t)v[i];
        hi[i] = h;
        lo[i] = (bf16_t)(v[i] - (float)h);
    }
}

// Same, with a channel-wise f32 scale vector s (same K layout as p)
__device__ __forceinline__ void split_a_scaled(const float* __restrict__ p,
                                               const float* __restrict__ s,
                                               v16bf& hi, v16bf& lo) {
    v4f q0 = *(const v4f*)(p),     s0 = *(const v4f*)(s);
    v4f q1 = *(const v4f*)(p + 4), s1 = *(const v4f*)(s + 4);
    v4f q2 = *(const v4f*)(p + 16), s2 = *(const v4f*)(s + 16);
    v4f q3 = *(const v4f*)(p + 20), s3 = *(const v4f*)(s + 20);
    float v[16];
#pragma unroll
    for (int i = 0; i < 4; ++i) {
        v[i]      = q0[i] * s0[i];
        v[4 + i]  = q1[i] * s1[i];
        v[8 + i]  = q2[i] * s2[i];
        v[12 + i] = q3[i] * s3[i];
    }
#pragma unroll
    for (int i = 0; i < 16; ++i) {
        bf16_t h = (bf16_t)v[i];
        hi[i] = h;
        lo[i] = (bf16_t)(v[i] - (float)h);
    }
}

__device__ __forceinline__ float wave_sum(float v) {
#pragma unroll
    for (int off = 16; off > 0; off >>= 1) v += __shfl_xor(v, off, 32);
    return v;
}

// ---------------------------------------------------------------------------
// Kernel: zero accumulators
// ---------------------------------------------------------------------------
__global__ void k_zero(float* __restrict__ p, int n) {
    int i = blockIdx.x * blockDim.x + threadIdx.x;
    if (i < n) p[i] = 0.0f;
}

// ---------------------------------------------------------------------------
// Kernel: split f32 matrix -> bf16 hi + lo  (weights; L2-resident afterwards)
// ---------------------------------------------------------------------------
__global__ void k_split(const float* __restrict__ src, bf16_t* __restrict__ hi,
                        bf16_t* __restrict__ lo, int n) {
    int i = blockIdx.x * blockDim.x + threadIdx.x;
    if (i < n) {
        float v = src[i];
        bf16_t h = (bf16_t)v;
        hi[i] = h;
        lo[i] = (bf16_t)(v - (float)h);
    }
}

// ---------------------------------------------------------------------------
// Kernel 1 (fused stage 1): block = 16 tokens.
//   kv(16x2048) = x_tile . Wkv^T  (split-bf16 WMMA, f32 accum) -> LDS
//   gate[t] = sum_h dot(k_h/||k_h||, v_h)
//   xw = (x/||x||)*gate*scaler + bias   -> stored pre-split (hi/lo bf16)
// ---------------------------------------------------------------------------
__global__ void k_stage1(const float* __restrict__ x,
                         const unsigned char* __restrict__ mask,
                         const bf16_t* __restrict__ wkvh,
                         const bf16_t* __restrict__ wkvl,
                         const float* __restrict__ scaler,
                         const float* __restrict__ bias,
                         bf16_t* __restrict__ xwh,
                         bf16_t* __restrict__ xwl) {
    extern __shared__ float sm[];               // 16 x 2048 floats = 128 KB
    const int tid  = threadIdx.x;
    const int lane = tid & 31;
    const int wave = tid >> 5;                  // 8 waves
    const int l    = lane & 15;
    const int kh   = lane >> 4;
    const int row0 = blockIdx.x * 16;

    const float* arow = x + (size_t)(row0 + l) * D_DIM + 8 * kh;
    const v8f zero = {};

    for (int cc = 0; cc < 4; ++cc) {
        const int n0 = wave * 256 + cc * 64;
        const bf16_t* bh[4];
        const bf16_t* bl[4];
#pragma unroll
        for (int j = 0; j < 4; ++j) {
            bh[j] = wkvh + (size_t)(n0 + j * 16 + l) * D_DIM + 8 * kh;
            bl[j] = wkvl + (size_t)(n0 + j * 16 + l) * D_DIM + 8 * kh;
        }
        v8f acc[4];
#pragma unroll
        for (int j = 0; j < 4; ++j) acc[j] = zero;
        for (int k = 0; k < D_DIM; k += 32) {
            v16bf ahi, alo;
            split_a(arow + k, ahi, alo);
#pragma unroll
            for (int j = 0; j < 4; ++j) {
                v16bf bhi = load16(bh[j] + k);
                v16bf blo = load16(bl[j] + k);
                acc[j] = wmma_bf16(ahi, bhi, acc[j]);
                acc[j] = wmma_bf16(ahi, blo, acc[j]);
                acc[j] = wmma_bf16(alo, bhi, acc[j]);
            }
        }
        const int mrow = kh * 8;
#pragma unroll
        for (int j = 0; j < 4; ++j)
#pragma unroll
            for (int r = 0; r < 8; ++r)
                sm[(size_t)(mrow + r) * KV_DIM + n0 + j * 16 + l] = acc[j][r];
    }
    __syncthreads();

    // gate + xw: each wave handles 2 of the 16 tokens
    for (int t = wave; t < 16; t += 8) {
        const float* krow = sm + (size_t)t * KV_DIM;
        float gate = 0.0f;
#pragma unroll
        for (int h = 0; h < 8; ++h) {
            const float* kp = krow + h * 256 + lane * 4;
            const float* vp = kp + 128;
            float kk = 0.0f, kv = 0.0f;
#pragma unroll
            for (int i = 0; i < 4; ++i) {
                float kv_ = kp[i], vv_ = vp[i];
                kk += kv_ * kv_;
                kv += kv_ * vv_;
            }
            kk = wave_sum(kk);
            kv = wave_sum(kv);
            gate += kv * rsqrtf(kk);
        }
        const int grow = row0 + t;
        const float* xr = x + (size_t)grow * D_DIM;
        float xs = 0.0f;
        for (int c = lane; c < D_DIM; c += 32) { float v = xr[c]; xs += v * v; }
        xs = wave_sum(xs);
        const float scale = gate * rsqrtf(xs);
        const bool msk = (mask[grow] != 0);
        bf16_t* ohr = xwh + (size_t)grow * D_DIM;
        bf16_t* olr = xwl + (size_t)grow * D_DIM;
        for (int c = lane; c < D_DIM; c += 32) {
            float w = msk ? 0.0f : (xr[c] * scale * scaler[c] + bias[c]);
            bf16_t h = (bf16_t)w;
            ohr[c] = h;
            olr[c] = (bf16_t)(w - (float)h);
        }
    }
}

// ---------------------------------------------------------------------------
// Kernel 2: q = xw . Wq^T (stored f32), plus s_qq[b,d] += sum_n q^2
// Block: 32 rows x 256 cols; wave = 16x64 (4 accumulators)
// ---------------------------------------------------------------------------
__global__ void k_qgemm(const bf16_t* __restrict__ xwh,
                        const bf16_t* __restrict__ xwl,
                        const bf16_t* __restrict__ wqh,
                        const bf16_t* __restrict__ wql,
                        float* __restrict__ q,
                        float* __restrict__ s_qq) {
    const int tid  = threadIdx.x;
    const int lane = tid & 31;
    const int wave = tid >> 5;
    const int l    = lane & 15;
    const int kh   = lane >> 4;
    const int m0   = blockIdx.x * 32 + (wave >> 2) * 16;
    const int n0   = blockIdx.y * 256 + (wave & 3) * 64;
    const int bidx = (blockIdx.x * 32) / N_TOK;

    const bf16_t* ah = xwh + (size_t)(m0 + l) * D_DIM + 8 * kh;
    const bf16_t* al = xwl + (size_t)(m0 + l) * D_DIM + 8 * kh;
    const bf16_t* bh[4];
    const bf16_t* bl[4];
#pragma unroll
    for (int j = 0; j < 4; ++j) {
        bh[j] = wqh + (size_t)(n0 + j * 16 + l) * D_DIM + 8 * kh;
        bl[j] = wql + (size_t)(n0 + j * 16 + l) * D_DIM + 8 * kh;
    }
    const v8f zero = {};
    v8f acc[4];
#pragma unroll
    for (int j = 0; j < 4; ++j) acc[j] = zero;
    for (int k = 0; k < D_DIM; k += 32) {
        v16bf ahi = load_a16(ah + k);
        v16bf alo = load_a16(al + k);
#pragma unroll
        for (int j = 0; j < 4; ++j) {
            v16bf bhi = load16(bh[j] + k);
            v16bf blo = load16(bl[j] + k);
            acc[j] = wmma_bf16(ahi, bhi, acc[j]);
            acc[j] = wmma_bf16(ahi, blo, acc[j]);
            acc[j] = wmma_bf16(alo, bhi, acc[j]);
        }
    }
    const int mrow = m0 + kh * 8;
#pragma unroll
    for (int j = 0; j < 4; ++j) {
        float cs = 0.0f;
#pragma unroll
        for (int r = 0; r < 8; ++r) {
            q[(size_t)(mrow + r) * D_DIM + n0 + j * 16 + l] = acc[j][r];
            cs += acc[j][r] * acc[j][r];
        }
        cs += __shfl_xor(cs, 16, 32);
        if (kh == 0) atomicAdd(&s_qq[bidx * D_DIM + n0 + j * 16 + l], cs);
    }
}

// ---------------------------------------------------------------------------
// Kernel 3: k2/v2 = xw . Wkv^T (never stored); accumulate
//   s_kk[b,d] += sum_n k2^2 ; s_kv[b,d] += sum_n k2*v2
// ---------------------------------------------------------------------------
__global__ void k_kvstats(const bf16_t* __restrict__ xwh,
                          const bf16_t* __restrict__ xwl,
                          const bf16_t* __restrict__ wkvh,
                          const bf16_t* __restrict__ wkvl,
                          float* __restrict__ s_kk,
                          float* __restrict__ s_kv) {
    const int tid  = threadIdx.x;
    const int lane = tid & 31;
    const int wave = tid >> 5;
    const int l    = lane & 15;
    const int kh   = lane >> 4;
    const int m0   = blockIdx.x * 32 + (wave >> 2) * 16;
    const int n0   = blockIdx.y * 256 + (wave & 3) * 64;
    const int bidx = (blockIdx.x * 32) / N_TOK;

    const bf16_t* ah = xwh + (size_t)(m0 + l) * D_DIM + 8 * kh;
    const bf16_t* al = xwl + (size_t)(m0 + l) * D_DIM + 8 * kh;
    const bf16_t* bkh[4];
    const bf16_t* bkl[4];
    const bf16_t* bvh[4];
    const bf16_t* bvl[4];
#pragma unroll
    for (int j = 0; j < 4; ++j) {
        bkh[j] = wkvh + (size_t)(n0 + j * 16 + l) * D_DIM + 8 * kh;
        bkl[j] = wkvl + (size_t)(n0 + j * 16 + l) * D_DIM + 8 * kh;
        bvh[j] = wkvh + (size_t)(D_DIM + n0 + j * 16 + l) * D_DIM + 8 * kh;
        bvl[j] = wkvl + (size_t)(D_DIM + n0 + j * 16 + l) * D_DIM + 8 * kh;
    }
    const v8f zero = {};
    v8f ak[4], av[4];
#pragma unroll
    for (int j = 0; j < 4; ++j) { ak[j] = zero; av[j] = zero; }
    for (int k = 0; k < D_DIM; k += 32) {
        v16bf ahi = load_a16(ah + k);
        v16bf alo = load_a16(al + k);
#pragma unroll
        for (int j = 0; j < 4; ++j) {
            v16bf bh = load16(bkh[j] + k);
            v16bf bb = load16(bkl[j] + k);
            ak[j] = wmma_bf16(ahi, bh, ak[j]);
            ak[j] = wmma_bf16(ahi, bb, ak[j]);
            ak[j] = wmma_bf16(alo, bh, ak[j]);
            v16bf ch = load16(bvh[j] + k);
            v16bf cl = load16(bvl[j] + k);
            av[j] = wmma_bf16(ahi, ch, av[j]);
            av[j] = wmma_bf16(ahi, cl, av[j]);
            av[j] = wmma_bf16(alo, ch, av[j]);
        }
    }
#pragma unroll
    for (int j = 0; j < 4; ++j) {
        float kk = 0.0f, kv = 0.0f;
#pragma unroll
        for (int r = 0; r < 8; ++r) {
            kk += ak[j][r] * ak[j][r];
            kv += ak[j][r] * av[j][r];
        }
        kk += __shfl_xor(kk, 16, 32);
        kv += __shfl_xor(kv, 16, 32);
        if (kh == 0) {
            atomicAdd(&s_kk[bidx * D_DIM + n0 + j * 16 + l], kk);
            atomicAdd(&s_kv[bidx * D_DIM + n0 + j * 16 + l], kv);
        }
    }
}

// ---------------------------------------------------------------------------
// Kernel 4: f[b,d] = s_kv / (sqrt(s_kk) * sqrt(s_qq))
// ---------------------------------------------------------------------------
__global__ void k_combine(const float* __restrict__ s_qq,
                          const float* __restrict__ s_kk,
                          const float* __restrict__ s_kv,
                          float* __restrict__ f) {
    int i = blockIdx.x * blockDim.x + threadIdx.x;
    if (i < B_SZ * D_DIM)
        f[i] = s_kv[i] * rsqrtf(s_kk[i]) * rsqrtf(s_qq[i]);
}

// ---------------------------------------------------------------------------
// Kernel 5: out = (q * f[b,:]) . Wout^T  (q scaled + split on the fly)
// ---------------------------------------------------------------------------
__global__ void k_out(const float* __restrict__ q,
                      const float* __restrict__ f,
                      const bf16_t* __restrict__ woh,
                      const bf16_t* __restrict__ wol,
                      float* __restrict__ out) {
    const int tid  = threadIdx.x;
    const int lane = tid & 31;
    const int wave = tid >> 5;
    const int l    = lane & 15;
    const int kh   = lane >> 4;
    const int m0   = blockIdx.x * 32 + (wave >> 2) * 16;
    const int n0   = blockIdx.y * 256 + (wave & 3) * 64;
    const int bidx = (blockIdx.x * 32) / N_TOK;

    const float* arow = q + (size_t)(m0 + l) * D_DIM + 8 * kh;
    const float* frow = f + (size_t)bidx * D_DIM + 8 * kh;
    const bf16_t* bh[4];
    const bf16_t* bl[4];
#pragma unroll
    for (int j = 0; j < 4; ++j) {
        bh[j] = woh + (size_t)(n0 + j * 16 + l) * D_DIM + 8 * kh;
        bl[j] = wol + (size_t)(n0 + j * 16 + l) * D_DIM + 8 * kh;
    }
    const v8f zero = {};
    v8f acc[4];
#pragma unroll
    for (int j = 0; j < 4; ++j) acc[j] = zero;
    for (int k = 0; k < D_DIM; k += 32) {
        v16bf ahi, alo;
        split_a_scaled(arow + k, frow + k, ahi, alo);
#pragma unroll
        for (int j = 0; j < 4; ++j) {
            v16bf bhi = load16(bh[j] + k);
            v16bf blo = load16(bl[j] + k);
            acc[j] = wmma_bf16(ahi, bhi, acc[j]);
            acc[j] = wmma_bf16(ahi, blo, acc[j]);
            acc[j] = wmma_bf16(alo, bhi, acc[j]);
        }
    }
    const int mrow = m0 + kh * 8;
#pragma unroll
    for (int j = 0; j < 4; ++j)
#pragma unroll
        for (int r = 0; r < 8; ++r)
            out[(size_t)(mrow + r) * D_DIM + n0 + j * 16 + l] = acc[j][r];
}

// ---------------------------------------------------------------------------
// Host launch
// ---------------------------------------------------------------------------
extern "C" void kernel_launch(void* const* d_in, const int* in_sizes, int n_in,
                              void* d_out, int out_size, void* d_ws, size_t ws_size,
                              hipStream_t stream) {
    const float*         x      = (const float*)d_in[0];
    const unsigned char* mask   = (const unsigned char*)d_in[1];
    const float*         Wq     = (const float*)d_in[2];
    const float*         Wkv    = (const float*)d_in[3];
    const float*         scaler = (const float*)d_in[4];
    const float*         bias   = (const float*)d_in[5];
    const float*         Wout   = (const float*)d_in[6];
    float*               outp   = (float*)d_out;

    char* ws = (char*)d_ws;
    const size_t nx   = (size_t)B_SZ * N_TOK * D_DIM;   // 16M elements
    const size_t nkv  = (size_t)KV_DIM * D_DIM;         // 2M
    const size_t nsq  = (size_t)D_DIM * D_DIM;          // 1M
    size_t off = 0;
    bf16_t* xwh  = (bf16_t*)(ws + off); off += nx * 2;          // 32 MB
    bf16_t* xwl  = (bf16_t*)(ws + off); off += nx * 2;          // 32 MB
    float*  qbuf = (float*)(ws + off);  off += nx * 4;          // 64 MB
    bf16_t* wkvh = (bf16_t*)(ws + off); off += nkv * 2;         //  4 MB
    bf16_t* wkvl = (bf16_t*)(ws + off); off += nkv * 2;         //  4 MB
    bf16_t* wqh  = (bf16_t*)(ws + off); off += nsq * 2;         //  2 MB
    bf16_t* wql  = (bf16_t*)(ws + off); off += nsq * 2;         //  2 MB
    bf16_t* woh  = (bf16_t*)(ws + off); off += nsq * 2;         //  2 MB
    bf16_t* wol  = (bf16_t*)(ws + off); off += nsq * 2;         //  2 MB
    float*  stats = (float*)(ws + off);
    float*  s_qq = stats;
    float*  s_kk = stats + B_SZ * D_DIM;
    float*  s_kv = stats + 2 * B_SZ * D_DIM;
    float*  fbuf = stats + 3 * B_SZ * D_DIM;

    const int nstats = 3 * B_SZ * D_DIM;
    k_zero<<<dim3((nstats + 255) / 256), dim3(256), 0, stream>>>(s_qq, nstats);

    // Split weights into bf16 hi/lo (16 MB total; stays L2-resident)
    k_split<<<dim3((int)((nkv + 255) / 256)), dim3(256), 0, stream>>>(Wkv, wkvh, wkvl, (int)nkv);
    k_split<<<dim3((int)((nsq + 255) / 256)), dim3(256), 0, stream>>>(Wq,  wqh,  wql,  (int)nsq);
    k_split<<<dim3((int)((nsq + 255) / 256)), dim3(256), 0, stream>>>(Wout, woh, wol,  (int)nsq);

    k_stage1<<<dim3((B_SZ * N_TOK) / 16), dim3(256),
               (size_t)16 * KV_DIM * sizeof(float), stream>>>(
        x, mask, wkvh, wkvl, scaler, bias, xwh, xwl);

    dim3 gemm_grid((B_SZ * N_TOK) / 32, D_DIM / 256);
    k_qgemm  <<<gemm_grid, dim3(256), 0, stream>>>(xwh, xwl, wqh, wql, qbuf, s_qq);
    k_kvstats<<<gemm_grid, dim3(256), 0, stream>>>(xwh, xwl, wkvh, wkvl, s_kk, s_kv);

    k_combine<<<dim3((B_SZ * D_DIM + 255) / 256), dim3(256), 0, stream>>>(
        s_qq, s_kk, s_kv, fbuf);

    k_out<<<gemm_grid, dim3(256), 0, stream>>>(qbuf, fbuf, woh, wol, outp);
}
